// HierarchicalBert_31207232373315
// MI455X (gfx1250) — matrix operand order repaired and checked by reading, not compiled, not run on hardware
//
#include <hip/hip_runtime.h>

// ---------------------------------------------------------------------------
// Types for CDNA5 WMMA (wave32): v_wmma_f32_16x16x32_bf16
// ---------------------------------------------------------------------------
typedef __attribute__((ext_vector_type(16))) __bf16 v16bf;
typedef __attribute__((ext_vector_type(8)))  float  v8f;

struct U16x16 { unsigned short u[16]; };
struct U4x2   { uint4 a, b; };

__device__ __forceinline__ unsigned short f2bf(float f) {
  unsigned int u = __builtin_bit_cast(unsigned int, f);
  u += 0x7fffu + ((u >> 16) & 1u);      // round-to-nearest-even
  return (unsigned short)(u >> 16);
}

// model constants
#define Dd   768
#define Hh   12
#define DHh  64
#define FFd  3072
#define Vv   32000
#define Bb   4
#define SEQ_ 512
#define NSEG 2
#define TT   4

#define FLAG_BIAS  1
#define FLAG_RES   2
#define FLAG_SILU  4
#define FLAG_OUTBF 8

// ---------------------------------------------------------------------------
// Weight convert+transpose: W[K,N] f32 -> Wt[N,K] bf16
// ---------------------------------------------------------------------------
__global__ void convtr_kernel(const float* __restrict__ W, unsigned short* __restrict__ Wt,
                              int K, int N) {
  size_t idx = (size_t)blockIdx.x * blockDim.x + threadIdx.x;
  if (idx >= (size_t)K * N) return;
  size_t n = idx / K, k = idx % K;
  Wt[idx] = f2bf(W[k * (size_t)N + n]);
}

// ---------------------------------------------------------------------------
// GEMM: C[M,N] = epilogue(A[M,K] @ W[K,N]); A and Wt both bf16 ([M,K] / [N,K]).
// 256 threads = 8 waves; tile 64(M) x 128(N), BK=32; each wave 32x32 via 2x2
// WMMA. Double-buffered LDS filled with GLOBAL_LOAD_ASYNC_TO_LDS_B128
// (ASYNCcnt), synced with s_wait_asynccnt + barrier. Layouts per ISA 7.12.2.
// ---------------------------------------------------------------------------
#define BM 64
#define BN 128
#define BK 32

__global__ __launch_bounds__(256) void gemm_bf16_kernel(
    const unsigned short* __restrict__ Abf, const unsigned short* __restrict__ Wt,
    const float* __restrict__ bias, const float* residual, void* Cout,
    int M, int N, int K, int flags)
{
  __shared__ unsigned short Ash[2][BM][BK];
  __shared__ unsigned short Bsh[2][BN][BK];
  const int tid  = threadIdx.x;
  const int lane = tid & 31;
  const int wave = tid >> 5;
  const int half = lane >> 4;
  const int l15  = lane & 15;
  const int m0 = blockIdx.y * BM;
  const int n0 = blockIdx.x * BN;
  const int mw = (wave >> 2) * 32;   // 0 / 32
  const int nw = (wave & 3) * 32;    // 0..96

  // per-thread staging coordinates
  const int ar = tid >> 2, ac = (tid & 3) * 8;    // A: 64x32, 16B per thread
  const int br = tid >> 1, bc = (tid & 1) * 16;   // B: 128x32, 32B per thread
  const unsigned long long agbase = (unsigned long long)(Abf + (size_t)(m0 + ar) * K + ac);
  const unsigned long long bgbase = (unsigned long long)(Wt + (size_t)(n0 + br) * K + bc);

  auto stage = [&](int k0, int buf) {
    unsigned long long ag = agbase + (unsigned long long)k0 * 2ull;
    unsigned long long bg = bgbase + (unsigned long long)k0 * 2ull;
    unsigned alds = (unsigned)(size_t)&Ash[buf][ar][ac];
    unsigned blds = (unsigned)(size_t)&Bsh[buf][br][bc];
    asm volatile("global_load_async_to_lds_b128 %0, %1, off"
                 :: "v"(alds), "v"(ag) : "memory");
    asm volatile("global_load_async_to_lds_b128 %0, %1, off\n\t"
                 "global_load_async_to_lds_b128 %2, %3, off"
                 :: "v"(blds), "v"(bg), "v"(blds + 16u), "v"(bg + 32ull) : "memory");
    if (k0 + 2 * BK < K)   // prefetch the K+2 tile of the streamed weights
      __builtin_prefetch((const void*)(Wt + (size_t)(n0 + br) * K + bc + k0 + 2 * BK), 0, 1);
  };

  v8f acc[2][2] = {};
  const int nk = K / BK;
  stage(0, 0);
  for (int s = 0; s < nk; ++s) {
    const int buf = s & 1;
    asm volatile("s_wait_asynccnt 0x0" ::: "memory");
    __syncthreads();
    if (s + 1 < nk) stage((s + 1) * BK, buf ^ 1);

    // A frags: lane holds row l15; elems 0..7 = K[koff..], 8..15 = K[16+koff..]
    v16bf afrag[2], bfrag[2];
    const int koffA  = half * 8;
    const int kbaseB = half * 8;
    for (int i = 0; i < 2; ++i) {
      int r = mw + i * 16 + l15;
      U4x2 t;
      t.a = *(const uint4*)&Ash[buf][r][koffA];
      t.b = *(const uint4*)&Ash[buf][r][16 + koffA];
      afrag[i] = __builtin_bit_cast(v16bf, t);
    }
    for (int j = 0; j < 2; ++j) {
      int n = nw + j * 16 + l15;
      U4x2 t;
      t.a = *(const uint4*)&Bsh[buf][n][kbaseB];
      t.b = *(const uint4*)&Bsh[buf][n][16 + kbaseB];
      bfrag[j] = __builtin_bit_cast(v16bf, t);
    }
    for (int i = 0; i < 2; ++i)
      for (int j = 0; j < 2; ++j)
        acc[i][j] = __builtin_amdgcn_wmma_f32_16x16x32_bf16(
            false, afrag[i], false, bfrag[j], (short)0, acc[i][j], false, false);
  }

  // epilogue: D layout — elem r -> row = base + half*8 + r, col = l15
  for (int i = 0; i < 2; ++i)
    for (int j = 0; j < 2; ++j) {
      int col = n0 + nw + j * 16 + l15;
      if (col >= N) continue;
      int rbase = m0 + mw + i * 16 + half * 8;
      float bv = (flags & FLAG_BIAS) ? bias[col] : 0.f;
      for (int r = 0; r < 8; ++r) {
        int row = rbase + r;
        if (row >= M) break;
        float v = acc[i][j][r] + bv;
        if (flags & FLAG_SILU) v = v / (1.f + __expf(-v));
        if (flags & FLAG_RES)  v += residual[(size_t)row * N + col];
        if (flags & FLAG_OUTBF)
          ((unsigned short*)Cout)[(size_t)row * N + col] = f2bf(v);
        else
          ((float*)Cout)[(size_t)row * N + col] = v;
      }
    }
}

// ---------------------------------------------------------------------------
// Flash-style attention: one wave per (b, h, 16-row q-tile). Inputs bf16.
// S = Q K^T via 2 bf16 WMMAs (d=64), online softmax, O += P V via 4 WMMAs.
// ---------------------------------------------------------------------------
__global__ __launch_bounds__(32) void attn_kernel(
    const unsigned short* __restrict__ Q, const unsigned short* __restrict__ K,
    const unsigned short* __restrict__ V, unsigned short* __restrict__ O,
    const float* __restrict__ amask, int seq, int isglobal, int win, float scale)
{
  __shared__ unsigned short Psh[16][32];
  __shared__ unsigned short Vsh[64][32];
  const int lane = threadIdx.x;
  const int half = lane >> 4, l15 = lane & 15;
  const int qt = blockIdx.x;
  const int bh = blockIdx.y;
  const int b = bh / Hh, h = bh % Hh;
  const size_t base = (size_t)b * seq * Dd + (size_t)h * DHh;
  const float NEG = -__builtin_inff();

  // Q fragments (two A frags covering d=0..31 and d=32..63)
  v16bf qf[2];
  {
    int qrow = qt * 16 + l15;
    int qc = (qrow < seq) ? qrow : 0;
    const unsigned short* qp = Q + base + (size_t)qc * Dd;
    for (int f = 0; f < 2; ++f) {
      U4x2 t;
      t.a = *(const uint4*)(qp + f * 32 + half * 8);
      t.b = *(const uint4*)(qp + f * 32 + 16 + half * 8);
      qf[f] = __builtin_bit_cast(v16bf, t);
    }
  }

  float m[8], l[8];
  for (int r = 0; r < 8; ++r) { m[r] = NEG; l[r] = 0.f; }
  v8f oacc[4] = {};

  const int nkt = (seq + 15) / 16;
  for (int kk = 0; kk < nkt; kk += 2) {
    float sraw[2][8];
    float tmax[8];
    for (int r = 0; r < 8; ++r) tmax[r] = NEG;

    for (int t = 0; t < 2; ++t) {
      int k0 = (kk + t) * 16;
      int kcol = k0 + l15;
      bool kvalid = (kcol < seq);
      const unsigned short* kp = K + base + (size_t)(kvalid ? kcol : 0) * Dd;
      v8f s = {};
      for (int f = 0; f < 2; ++f) {
        // B frag of K^T: elems 0..7 = K[kcol][f*32 + half*8 ..], 8..15 = K[kcol][f*32 + 16 + half*8 ..]
        U4x2 tb;
        tb.a = *(const uint4*)(kp + f * 32 + half * 8);
        tb.b = *(const uint4*)(kp + f * 32 + 16 + half * 8);
        v16bf bf = __builtin_bit_cast(v16bf, tb);
        s = __builtin_amdgcn_wmma_f32_16x16x32_bf16(
            false, qf[f], false, bf, (short)0, s, false, false);
      }
      float pad = 0.f;
      if (amask && kvalid) pad = (1.f - amask[(size_t)b * seq + kcol]) * -1e9f;
      for (int r = 0; r < 8; ++r) {
        int qr = qt * 16 + r + half * 8;
        float v = s[r] * scale + pad;
        if (!kvalid) v = NEG;
        if (!isglobal) {
          int diff = kcol - qr;
          if (diff < -win || diff > win) v = NEG;
        }
        sraw[t][r] = v;
        tmax[r] = fmaxf(tmax[r], v);
      }
    }
    // row-wide max across the 16 lanes of each half (xor masks < 16 stay in-half)
    for (int r = 0; r < 8; ++r) {
      float v = tmax[r];
      for (int off = 1; off < 16; off <<= 1) v = fmaxf(v, __shfl_xor(v, off, 32));
      tmax[r] = v;
    }
    float alpha[8];
    for (int r = 0; r < 8; ++r) {
      float mn = fmaxf(m[r], tmax[r]);
      float a = __expf(m[r] - mn);
      if (!(a == a)) a = 1.f;   // -inf - -inf
      alpha[r] = a;
      m[r] = mn;
    }
    float p[2][8], psum[8];
    for (int r = 0; r < 8; ++r) psum[r] = 0.f;
    for (int t = 0; t < 2; ++t)
      for (int r = 0; r < 8; ++r) {
        float d = sraw[t][r] - m[r];
        float pv = (d == d) ? __expf(d) : 0.f;
        p[t][r] = pv;
        psum[r] += pv;
      }
    for (int r = 0; r < 8; ++r) {
      float v = psum[r];
      for (int off = 1; off < 16; off <<= 1) v += __shfl_xor(v, off, 32);
      l[r] = l[r] * alpha[r] + v;
    }
    for (int j = 0; j < 4; ++j)
      for (int r = 0; r < 8; ++r) oacc[j][r] *= alpha[r];

    __syncthreads();
    // transpose P (D layout) into A-fragment layout via LDS
    for (int t = 0; t < 2; ++t)
      for (int r = 0; r < 8; ++r)
        Psh[r + half * 8][t * 16 + l15] = f2bf(p[t][r]);
    // stage V (32 k-rows x 64 d, bf16) transposed into Vsh[d][k]
    {
      int krow = kk * 16 + lane;
      if (krow < seq) {
        const uint4* vp = (const uint4*)(V + base + (size_t)krow * Dd);
        for (int i = 0; i < 8; ++i) {
          uint4 w = vp[i];
          unsigned vals[4] = {w.x, w.y, w.z, w.w};
          for (int j = 0; j < 4; ++j) {
            Vsh[i * 8 + j * 2 + 0][lane] = (unsigned short)(vals[j] & 0xffffu);
            Vsh[i * 8 + j * 2 + 1][lane] = (unsigned short)(vals[j] >> 16);
          }
        }
      } else {
        for (int d = 0; d < 64; ++d) Vsh[d][lane] = 0;
      }
    }
    __syncthreads();

    v16bf pf;
    {
      int koff = half * 8;
      U16x16 t;
      for (int j = 0; j < 8; ++j) t.u[j]     = Psh[l15][koff + j];
      for (int j = 0; j < 8; ++j) t.u[8 + j] = Psh[l15][16 + koff + j];
      pf = __builtin_bit_cast(v16bf, t);
    }
    for (int j = 0; j < 4; ++j) {
      int d = j * 16 + l15;
      U16x16 tb;
      for (int e = 0; e < 16; ++e) tb.u[e] = Vsh[d][(e < 8 ? half * 8 + e : 16 + half * 8 + (e - 8))];
      v16bf vf = __builtin_bit_cast(v16bf, tb);
      oacc[j] = __builtin_amdgcn_wmma_f32_16x16x32_bf16(
          false, pf, false, vf, (short)0, oacc[j], false, false);
    }
    __syncthreads();
  }

  for (int r = 0; r < 8; ++r) {
    int qr = qt * 16 + r + half * 8;
    if (qr >= seq) continue;
    float inv = (l[r] > 0.f) ? 1.f / l[r] : 0.f;
    for (int j = 0; j < 4; ++j)
      O[base + (size_t)qr * Dd + j * 16 + l15] = f2bf(oacc[j][r] * inv);
  }
}

// ---------------------------------------------------------------------------
// Elementwise helpers
// ---------------------------------------------------------------------------
__global__ void embed_kernel(const int* __restrict__ ids, const int* __restrict__ tt,
                             const float* __restrict__ emb, const float* __restrict__ temb,
                             float* __restrict__ out, int total) {
  int idx = blockIdx.x * blockDim.x + threadIdx.x;
  if (idx >= total) return;
  int row = idx / Dd, d = idx % Dd;
  out[idx] = emb[(size_t)ids[row] * Dd + d] + temb[(size_t)tt[row] * Dd + d];
}

// RMSNorm per row; optional broadcast addend zadd[b*D+d]; obf: bf16 output
__global__ __launch_bounds__(256) void rms_kernel(
    const float* __restrict__ in, const float* __restrict__ zadd,
    const float* __restrict__ w, void* out, int seq, int obf) {
  __shared__ float red[256];
  int row = blockIdx.x;
  int b = row / seq;
  float ss = 0.f;
  for (int d = threadIdx.x; d < Dd; d += 256) {
    float v = in[(size_t)row * Dd + d];
    if (zadd) v += zadd[(size_t)b * Dd + d];
    ss += v * v;
  }
  red[threadIdx.x] = ss;
  __syncthreads();
  for (int s = 128; s > 0; s >>= 1) {
    if (threadIdx.x < s) red[threadIdx.x] += red[threadIdx.x + s];
    __syncthreads();
  }
  float rs = rsqrtf(red[0] / Dd + 1e-6f);
  for (int d = threadIdx.x; d < Dd; d += 256) {
    float v = in[(size_t)row * Dd + d];
    if (zadd) v += zadd[(size_t)b * Dd + d];
    v = v * rs * w[d];
    if (obf) ((unsigned short*)out)[(size_t)row * Dd + d] = f2bf(v);
    else     ((float*)out)[(size_t)row * Dd + d] = v;
  }
}

// RoPE (per reference): out[j]=x[j]*cos + rot[j]*sin per 64-d head; bf16 out
__global__ void rope_kernel(const float* __restrict__ src, unsigned short* __restrict__ dst,
                            int total, int seq, float base) {
  int idx = blockIdx.x * blockDim.x + threadIdx.x;
  if (idx >= total) return;
  int row = idx / Dd, col = idx % Dd;
  int dj = col & 63;
  int pos = row % seq;
  int i = dj & 31;
  float ang = pos * __powf(base, -(float)i / 32.f);
  float c = __cosf(ang), s = __sinf(ang);
  const float* x = src + (size_t)row * Dd + (col & ~63);
  float rot = (dj < 32) ? -x[2 * dj + 1] : x[2 * (dj - 32)];
  dst[idx] = f2bf(x[dj] * c + rot * s);
}

// h_low = (x?) + zL + broadcast(zH)
__global__ void seed_low_kernel(const float* __restrict__ x, const float* __restrict__ zL,
                                const float* __restrict__ zH, float* __restrict__ out,
                                int total, int seq) {
  int idx = blockIdx.x * blockDim.x + threadIdx.x;
  if (idx >= total) return;
  int row = idx / Dd, d = idx % Dd;
  int b = row / seq;
  float v = zL[idx] + zH[(size_t)b * Dd + d];
  if (x) v += x[idx];
  out[idx] = v;
}

__global__ void bcast_kernel(const float* __restrict__ src, float* __restrict__ dst,
                             int total) {
  int idx = blockIdx.x * blockDim.x + threadIdx.x;
  if (idx >= total) return;
  dst[idx] = src[idx % Dd];
}

__global__ void copy_kernel(const float* __restrict__ src, float* __restrict__ dst, int n) {
  int idx = blockIdx.x * blockDim.x + threadIdx.x;
  if (idx < n) dst[idx] = src[idx];
}

// hH = zH + mean_n(zL)
__global__ void mean_seed_kernel(const float* __restrict__ zL, const float* __restrict__ zH,
                                 float* __restrict__ hH, int seq) {
  int idx = blockIdx.x * blockDim.x + threadIdx.x;
  if (idx >= Bb * Dd) return;
  int b = idx / Dd, d = idx % Dd;
  float s = 0.f;
  for (int n = 0; n < seq; ++n) s += zL[((size_t)b * seq + n) * Dd + d];
  hH[idx] = zH[idx] + s / seq;
}

__global__ void qhead_kernel(const float* __restrict__ hH, const float* __restrict__ qW,
                             const float* __restrict__ qb, float* __restrict__ out) {
  int idx = threadIdx.x;
  if (idx >= Bb * 2) return;
  int b = idx / 2, c = idx % 2;
  float s = qb[c];
  for (int d = 0; d < Dd; ++d) s += hH[(size_t)b * Dd + d] * qW[(size_t)d * 2 + c];
  out[idx] = s;
}

// ---------------------------------------------------------------------------
// Host orchestration
// ---------------------------------------------------------------------------
struct Blk {
  const float *attn_w, *Wq, *Wk, *Wv, *Wo, *bo, *ffn_w, *W1, *b1, *W2, *b2;
};

extern "C" void kernel_launch(void* const* d_in, const int* in_sizes, int n_in,
                              void* d_out, int out_size, void* d_ws, size_t ws_size,
                              hipStream_t stream) {
  (void)in_sizes; (void)n_in; (void)out_size; (void)ws_size;
  const int*   ids        = (const int*)d_in[0];
  const int*   ttids      = (const int*)d_in[1];
  const float* amask      = (const float*)d_in[2];
  const float* embed      = (const float*)d_in[3];
  const float* type_embed = (const float*)d_in[4];
  const float* embed_norm = (const float*)d_in[5];
  const float* zL0        = (const float*)d_in[6];
  const float* zH0        = (const float*)d_in[7];
  const float* final_norm = (const float*)d_in[8];
  const float* mlmW       = (const float*)d_in[9];
  const float* mlmb       = (const float*)d_in[10];
  const float* qW         = (const float*)d_in[11];
  const float* qb         = (const float*)d_in[12];

  Blk blk[8];
  for (int i = 0; i < 8; ++i) {
    int base = 13 + i * 11;
    blk[i].attn_w = (const float*)d_in[base + 0];
    blk[i].Wq     = (const float*)d_in[base + 1];
    blk[i].Wk     = (const float*)d_in[base + 2];
    blk[i].Wv     = (const float*)d_in[base + 3];
    blk[i].Wo     = (const float*)d_in[base + 4];
    blk[i].bo     = (const float*)d_in[base + 5];
    blk[i].ffn_w  = (const float*)d_in[base + 6];
    blk[i].W1     = (const float*)d_in[base + 7];
    blk[i].b1     = (const float*)d_in[base + 8];
    blk[i].W2     = (const float*)d_in[base + 9];
    blk[i].b2     = (const float*)d_in[base + 10];
  }
  float* out = (float*)d_out;

  // ---- workspace bump allocation ----
  char* cur = (char*)d_ws;
  auto alloc = [&](size_t bytes) -> void* {
    void* r = (void*)cur;
    cur += (bytes + 255) & ~(size_t)255;
    return r;
  };
  const size_t Mlow = (size_t)Bb * SEQ_;  // 2048
  unsigned short *WqT[8], *WkT[8], *WvT[8], *WoT[8], *W1T[8], *W2T[8];
  for (int i = 0; i < 8; ++i) {
    WqT[i] = (unsigned short*)alloc((size_t)Dd * Dd * 2);
    WkT[i] = (unsigned short*)alloc((size_t)Dd * Dd * 2);
    WvT[i] = (unsigned short*)alloc((size_t)Dd * Dd * 2);
    WoT[i] = (unsigned short*)alloc((size_t)Dd * Dd * 2);
    W1T[i] = (unsigned short*)alloc((size_t)Dd * FFd * 2);
    W2T[i] = (unsigned short*)alloc((size_t)FFd * Dd * 2);
  }
  unsigned short* mlmT = (unsigned short*)alloc((size_t)Dd * Vv * 2);

  // f32 residual-stream buffers
  float* xemb  = (float*)alloc(Mlow * Dd * 4);
  float* xraw  = (float*)alloc(Mlow * Dd * 4);
  float* zLbuf = (float*)alloc(Mlow * Dd * 4);
  float* zHbuf = (float*)alloc((size_t)Bb * Dd * 4);
  float* hHbuf = (float*)alloc((size_t)Bb * Dd * 4);
  float* qraw  = (float*)alloc(Mlow * Dd * 4);
  float* kraw  = (float*)alloc(Mlow * Dd * 4);
  // bf16 matrix-op operand buffers
  unsigned short* hnorm = (unsigned short*)alloc(Mlow * Dd * 2);
  unsigned short* qbf   = (unsigned short*)alloc(Mlow * Dd * 2);
  unsigned short* kbf   = (unsigned short*)alloc(Mlow * Dd * 2);
  unsigned short* vbf   = (unsigned short*)alloc(Mlow * Dd * 2);
  unsigned short* attno = (unsigned short*)alloc(Mlow * Dd * 2);
  unsigned short* ffbuf = (unsigned short*)alloc(Mlow * FFd * 2);
  unsigned short* onorm = (unsigned short*)alloc(Mlow * Dd * 2);

  auto convtr = [&](const float* W, unsigned short* Wt, int K, int N) {
    size_t tot = (size_t)K * N;
    convtr_kernel<<<(unsigned)((tot + 255) / 256), 256, 0, stream>>>(W, Wt, K, N);
  };
  for (int i = 0; i < 8; ++i) {
    convtr(blk[i].Wq, WqT[i], Dd, Dd);
    convtr(blk[i].Wk, WkT[i], Dd, Dd);
    convtr(blk[i].Wv, WvT[i], Dd, Dd);
    convtr(blk[i].Wo, WoT[i], Dd, Dd);
    convtr(blk[i].W1, W1T[i], Dd, FFd);
    convtr(blk[i].W2, W2T[i], FFd, Dd);
  }
  convtr(mlmW, mlmT, Dd, Vv);

  auto gemm = [&](const unsigned short* A, const unsigned short* Wt, const float* bias,
                  const float* res, void* C, int M, int N, int K, int flags) {
    dim3 g((N + BN - 1) / BN, (M + BM - 1) / BM);
    gemm_bf16_kernel<<<g, 256, 0, stream>>>(A, Wt, bias, res, C, M, N, K, flags);
  };

  // one transformer block, in-place on x[M,D] (f32 residual stream)
  auto run_block = [&](float* x, int M, int seq, int bi, bool isglobal, float rbase,
                       bool usemask) {
    int tot = M * Dd;
    rms_kernel<<<M, 256, 0, stream>>>(x, nullptr, blk[bi].attn_w, hnorm, seq, 1);
    gemm(hnorm, WqT[bi], nullptr, nullptr, qraw, M, Dd, Dd, 0);
    gemm(hnorm, WkT[bi], nullptr, nullptr, kraw, M, Dd, Dd, 0);
    gemm(hnorm, WvT[bi], nullptr, nullptr, vbf, M, Dd, Dd, FLAG_OUTBF);
    rope_kernel<<<(tot + 255) / 256, 256, 0, stream>>>(qraw, qbf, tot, seq, rbase);
    rope_kernel<<<(tot + 255) / 256, 256, 0, stream>>>(kraw, kbf, tot, seq, rbase);
    dim3 ag((seq + 15) / 16, (M / seq) * Hh);
    attn_kernel<<<ag, 32, 0, stream>>>(qbf, kbf, vbf, attno,
                                       usemask ? amask : nullptr, seq,
                                       isglobal ? 1 : 0, 64, 0.125f);
    gemm(attno, WoT[bi], blk[bi].bo, x, x, M, Dd, Dd, FLAG_BIAS | FLAG_RES);
    rms_kernel<<<M, 256, 0, stream>>>(x, nullptr, blk[bi].ffn_w, hnorm, seq, 1);
    gemm(hnorm, W1T[bi], blk[bi].b1, nullptr, ffbuf, M, FFd, Dd,
         FLAG_BIAS | FLAG_SILU | FLAG_OUTBF);
    gemm(ffbuf, W2T[bi], blk[bi].b2, x, x, M, Dd, FFd, FLAG_BIAS | FLAG_RES);
  };

  const int totLow = (int)(Mlow * Dd);
  // embeddings: x = rms(embed[ids]+type_embed[tt])  (kept f32, feeds residual)
  embed_kernel<<<(totLow + 255) / 256, 256, 0, stream>>>(ids, ttids, embed, type_embed,
                                                         xraw, totLow);
  rms_kernel<<<(int)Mlow, 256, 0, stream>>>(xraw, nullptr, embed_norm, xemb, SEQ_, 0);
  // init zL, zH from broadcast params
  bcast_kernel<<<(totLow + 255) / 256, 256, 0, stream>>>(zL0, zLbuf, totLow);
  bcast_kernel<<<(Bb * Dd + 255) / 256, 256, 0, stream>>>(zH0, zHbuf, Bb * Dd);

  for (int seg = 0; seg < NSEG; ++seg) {
    // h_low = (x if seg==0) + zL + zH  (in-place into zLbuf)
    seed_low_kernel<<<(totLow + 255) / 256, 256, 0, stream>>>(
        seg == 0 ? xemb : nullptr, zLbuf, zHbuf, zLbuf, totLow, SEQ_);
    for (int t = 0; t < TT; ++t)
      for (int i = 0; i < 4; ++i) {
        bool isg = ((i + 1) % 3 == 0);
        run_block(zLbuf, (int)Mlow, SEQ_, i, isg, isg ? 160000.f : 10000.f, true);
      }
    // zL = h_low (already in zLbuf). hH = zH + mean(zL)
    mean_seed_kernel<<<(Bb * Dd + 255) / 256, 256, 0, stream>>>(zLbuf, zHbuf, hHbuf, SEQ_);
    for (int i = 0; i < 4; ++i)
      run_block(hHbuf, Bb, 1, 4 + i, true, 160000.f, false);
    copy_kernel<<<(Bb * Dd + 255) / 256, 256, 0, stream>>>(hHbuf, zHbuf, Bb * Dd);
    // out = rms(zL + zH, final_norm); logits = out @ mlm_W + mlm_b
    rms_kernel<<<(int)Mlow, 256, 0, stream>>>(zLbuf, zHbuf, final_norm, onorm, SEQ_, 1);
    gemm(onorm, mlmT, mlmb, nullptr, out + (size_t)seg * Mlow * Vv,
         (int)Mlow, Vv, Dd, FLAG_BIAS);
    qhead_kernel<<<1, 64, 0, stream>>>(hHbuf, qW, qb,
                                       out + (size_t)NSEG * Mlow * Vv + (size_t)seg * Bb * 2);
  }
}
